// SynthesisStage_88923002896960
// MI455X (gfx1250) — matrix operand order repaired and checked by reading, not compiled, not run on hardware
//
#include <hip/hip_runtime.h>
#include <hip/hip_bf16.h>

typedef __bf16 bf16_t;
typedef __attribute__((ext_vector_type(16))) __bf16 v16bf;
typedef __attribute__((ext_vector_type(8)))  __bf16 v8bf;
typedef __attribute__((ext_vector_type(8)))  float  v8f;

#define B_    8
#define CI_   256
#define CO_   256
#define H_    128
#define W_    128
#define NBANK 4
#define XSTRIDE 40    // padded LDS row stride (bf16): 80B, 16B-aligned, bank-rotating
#define XCOLS   130   // 128 w + 1-column halo on each side

// wmod layout: [b][tap(9)][ciChunk(8)][co(256)][ci32(32)] (bf16)
// -> A tile for (b,tap,cc,coBase) is one contiguous 8KB block (async-DMA friendly).
__device__ __forceinline__ size_t wmod_index(int b, int k, int co, int ci) {
    return ((((size_t)(b * 9 + k) * 8 + (ci >> 5)) * CO_ + co) << 5) + (ci & 31);
}

// ---------------------------------------------------------------------------
// Kernel 1: adaptive kernel selection + modulation + demodulation.
// ---------------------------------------------------------------------------
__global__ __launch_bounds__(256)
void modulate_kernel(const float* __restrict__ weight,   // [N,Co,Ci,3,3]
                     const float* __restrict__ styles,   // [B,Ci]
                     const float* __restrict__ selector, // [B,N]
                     bf16_t* __restrict__ wmod)
{
    const int co = blockIdx.x;
    const int b  = blockIdx.y;
    const int tid = threadIdx.x;

    __shared__ float wbuf[CI_ * 9];
    __shared__ float red[256];

    float sel[NBANK];
#pragma unroll
    for (int n = 0; n < NBANK; ++n) sel[n] = selector[b * NBANK + n];

    float sumsq = 0.0f;
    for (int i = tid; i < CI_ * 9; i += 256) {
        const int ci = i / 9;
        const int k  = i - ci * 9;
        float acc = 0.0f;
#pragma unroll
        for (int n = 0; n < NBANK; ++n) {
            acc += sel[n] * weight[(((size_t)n * CO_ + co) * CI_ + ci) * 9 + k];
        }
        acc *= styles[b * CI_ + ci];
        wbuf[i] = acc;
        sumsq += acc * acc;
    }

    red[tid] = sumsq;
    __syncthreads();
#pragma unroll
    for (int off = 128; off > 0; off >>= 1) {
        if (tid < off) red[tid] += red[tid + off];
        __syncthreads();
    }
    const float dcoef = rsqrtf(red[0] + 1e-8f);

    for (int i = tid; i < CI_ * 9; i += 256) {
        const int ci = i / 9;
        const int k  = i - ci * 9;
        wmod[wmod_index(b, k, co, ci)] = (bf16_t)(wbuf[i] * dcoef);
    }
}

// ---------------------------------------------------------------------------
// Kernel 2: implicit-GEMM conv, bf16 WMMA, halo'd X rows staged once per
// ci-chunk, ping-pong async-DMA A tiles overlapping WMMA compute.
// Grid: (h=128, coBlock=2, b=8). Block: 256 threads = 8 waves.
// ---------------------------------------------------------------------------
__global__ __launch_bounds__(256)
void conv_wmma_kernel(const float*  __restrict__ x,     // [B,Ci,H,W]
                      const bf16_t* __restrict__ wmod,
                      const float*  __restrict__ noise, // [B,Co,H,W]
                      float*        __restrict__ out)   // [B,Co,H,W]
{
    const int h      = blockIdx.x;
    const int coBase = blockIdx.y * 128;
    const int b      = blockIdx.z;

    const int tid   = threadIdx.x;
    const int lane  = tid & 31;
    const int wid   = tid >> 5;
    const int waveM = wid & 3;   // co block of 32
    const int waveN = wid >> 2;  // w  block of 64

    __shared__ __align__(16) bf16_t Alds[2][128 * 32];       // ping-pong A tiles
    __shared__ __align__(16) bf16_t Xlds[3][XCOLS * XSTRIDE]; // 3 halo'd x rows

    v8f acc[2][4];
#pragma unroll
    for (int mi = 0; mi < 2; ++mi)
#pragma unroll
        for (int ni = 0; ni < 4; ++ni)
            acc[mi][ni] = (v8f){0.f, 0.f, 0.f, 0.f, 0.f, 0.f, 0.f, 0.f};

    const float* xb = x + (size_t)b * CI_ * H_ * W_;

    const int half = lane >> 4;   // ISA 7.12.2: K/M half select
    const int l16  = lane & 15;

    // Per-thread LDS byte addresses of this thread's two 16B async chunks, per buffer.
    const unsigned ldsA0[2] = { (unsigned)(size_t)&Alds[0][(size_t)tid * 8],
                                (unsigned)(size_t)&Alds[1][(size_t)tid * 8] };
    const unsigned ldsA1[2] = { (unsigned)(size_t)&Alds[0][(size_t)(tid + 256) * 8],
                                (unsigned)(size_t)&Alds[1][(size_t)(tid + 256) * 8] };

#pragma unroll 1
    for (int cc = 0; cc < 8; ++cc) {
        const int ciBase = cc * 32;

        auto issueA = [&](int kk, int p) {
            const bf16_t* wtile =
                wmod + ((((size_t)(b * 9 + kk) * 8 + cc) * CO_ + coBase) << 5);
            const bf16_t* g0 = wtile + (size_t)tid * 8;
            const bf16_t* g1 = g0 + 256 * 8;
            asm volatile("global_load_async_to_lds_b128 %0, %1, off"
                         :: "v"(ldsA0[p]), "v"(g0) : "memory");
            asm volatile("global_load_async_to_lds_b128 %0, %1, off"
                         :: "v"(ldsA1[p]), "v"(g1) : "memory");
        };

        __syncthreads();   // previous chunk fully consumed (X and A buffers)

        issueA(0, 0);      // A tile for tap 0 DMAs while we stage X

        // ---- stage 3 halo'd x rows for this ci-chunk ----
        {
            const int ciL = tid >> 3;                       // 0..31
            // zero halo columns 0 and 129 of each of the 3 rows
            if (tid < 96) {
                const int r  = tid >> 5;
                const int ci = tid & 31;
                Xlds[r][0 * XSTRIDE + ci]            = (bf16_t)0.0f;
                Xlds[r][(XCOLS - 1) * XSTRIDE + ci]  = (bf16_t)0.0f;
            }
#pragma unroll
            for (int r = 0; r < 3; ++r) {
                const int hIn = h + r - 1;
                const bool hOK = (hIn >= 0) && (hIn < H_);
                const float* xrow =
                    xb + ((size_t)(ciBase + ciL) * H_ + (hOK ? hIn : 0)) * W_;
#pragma unroll
                for (int j = 0; j < 4; ++j) {
                    const int w4 = ((tid & 7) + j * 8) * 4;  // aligned, coalesced
                    float4 v;
                    if (hOK) v = *(const float4*)(xrow + w4);
                    else     v = make_float4(0.f, 0.f, 0.f, 0.f);
                    const float vv[4] = {v.x, v.y, v.z, v.w};
#pragma unroll
                    for (int e = 0; e < 4; ++e)
                        Xlds[r][(w4 + e + 1) * XSTRIDE + ciL] = (bf16_t)vv[e];
                }
                if (cc < 7 && hOK)   // warm near caches for next ci-chunk
                    __builtin_prefetch(xrow + (size_t)32 * H_ * W_, 0, 1);
            }
        }

        asm volatile("s_wait_asynccnt 0x0" ::: "memory");   // A(tap0) landed
        __syncthreads();                                    // X + A0 visible to all

        // ---- tap loop: DMA next A while computing current ----
#pragma unroll 1
        for (int kk = 0; kk < 9; ++kk) {
            const int kh = kk / 3;
            const int kw = kk - kh * 3;
            const int p  = kk & 1;

            if (kk < 8) issueA(kk + 1, p ^ 1);  // buffer last read at tap kk-1

            // A fragments (ISA 7.12.2 16-bit A layout)
            v16bf afrag[2];
#pragma unroll
            for (int mi = 0; mi < 2; ++mi) {
                const int row = waveM * 32 + mi * 16 + l16;
                union { v16bf v; v8bf hh[2]; } u;
                u.hh[0] = *(const v8bf*)(&Alds[p][row * 32 + half * 8]);
                u.hh[1] = *(const v8bf*)(&Alds[p][row * 32 + 16 + half * 8]);
                afrag[mi] = u.v;
            }
            // B fragments: kw shift is a pure column offset into the halo'd rows
            v16bf bfrag[4];
#pragma unroll
            for (int ni = 0; ni < 4; ++ni) {
                const int col = waveN * 64 + ni * 16 + l16 + kw;   // 0..129
                union { v16bf v; v8bf hh[2]; } u;
                u.hh[0] = *(const v8bf*)(&Xlds[kh][col * XSTRIDE + half * 8]);
                u.hh[1] = *(const v8bf*)(&Xlds[kh][col * XSTRIDE + 16 + half * 8]);
                bfrag[ni] = u.v;
            }

#pragma unroll
            for (int mi = 0; mi < 2; ++mi)
#pragma unroll
                for (int ni = 0; ni < 4; ++ni)
                    acc[mi][ni] = __builtin_amdgcn_wmma_f32_16x16x32_bf16(
                        false, afrag[mi], false, bfrag[ni],
                        (short)0, acc[mi][ni], false, false);

            if (kk < 8) {
                asm volatile("s_wait_asynccnt 0x0" ::: "memory"); // own DMA landed
                __syncthreads();                                  // everyone's landed
            }
        }
    }

    // ---- epilogue: D layout (lane%16 = N; VGPR r -> M = 8*half + r) ----
#pragma unroll
    for (int mi = 0; mi < 2; ++mi) {
#pragma unroll
        for (int ni = 0; ni < 4; ++ni) {
            const int w = waveN * 64 + ni * 16 + l16;
#pragma unroll
            for (int r = 0; r < 8; ++r) {
                const int co = coBase + waveM * 32 + mi * 16 + half * 8 + r;
                const size_t idx = (((size_t)b * CO_ + co) * H_ + h) * W_ + w;
                out[idx] = acc[mi][ni][r] + noise[idx];
            }
        }
    }
}

// ---------------------------------------------------------------------------
extern "C" void kernel_launch(void* const* d_in, const int* in_sizes, int n_in,
                              void* d_out, int out_size, void* d_ws, size_t ws_size,
                              hipStream_t stream) {
    const float* x        = (const float*)d_in[0];
    const float* weight   = (const float*)d_in[1];
    const float* styles   = (const float*)d_in[2];
    const float* selector = (const float*)d_in[3];
    const float* noise    = (const float*)d_in[4];
    float*       out      = (float*)d_out;
    bf16_t*      wmod     = (bf16_t*)d_ws;   // B*9*Co*Ci*2 = 9.4 MB

    dim3 g1(CO_, B_);
    modulate_kernel<<<g1, 256, 0, stream>>>(weight, styles, selector, wmod);

    dim3 g2(H_, CO_ / 128, B_);
    conv_wmma_kernel<<<g2, 256, 0, stream>>>(x, wmod, noise, out);
}
